// ModelSample_67808943669901
// MI455X (gfx1250) — matrix operand order repaired and checked by reading, not compiled, not run on hardware
//
#include <hip/hip_runtime.h>
#include <math.h>

// ---------------------------------------------------------------------------
// FermiNet-style wavefunction for MI455X (gfx1250), wave32 + WMMA f16.
// One workgroup (8 wave32) per batch element. All GEMMs via
// v_wmma_f32_16x16x32_f16. Weights pre-transposed/converted to f16 (N x Kpad)
// in workspace so B fragments are two contiguous 16B global loads (L2-resident).
// Transcendentals use raw v_exp_f32 / v_log_f32 / v_rcp_f32 (branch-free,
// co-executable with XDL WMMA) instead of branchy ocml expansions.
// ---------------------------------------------------------------------------

typedef __attribute__((ext_vector_type(16))) _Float16 v16h;
typedef __attribute__((ext_vector_type(8)))  _Float16 v8h;
typedef __attribute__((ext_vector_type(8)))  float    v8f;

#define DEVFN static __device__ __forceinline__

// ---- problem constants ----
#define NE   16
#define NU   8
#define NSV  256
#define NPV  32
#define NA   4
#define NDET 8
#define KBIG 832        // 3*256 + 2*32, exactly 26 k-tiles of 32
#define KT_BIG 26

#define LOG2E_F  1.4426950408889634f
#define LN2_F    0.6931471805599453f

// branch-free fast math on raw gfx transcendentals
DEVFN float fast_exp(float x) {            // e^x
  return __builtin_amdgcn_exp2f(x * LOG2E_F);
}
DEVFN float fast_log(float x) {            // ln(x)
  return __builtin_amdgcn_logf(x) * LN2_F;
}
DEVFN float fast_tanh(float x) {
  // tanh(x) = 1 - 2/(exp(2x)+1); saturates correctly at +-inf, no branches
  float e = __builtin_amdgcn_exp2f(x * (2.0f * LOG2E_F));
  float r = __builtin_amdgcn_rcpf(e + 1.0f);
  return __builtin_fmaf(-2.0f, r, 1.0f);
}

// ---------------- shared memory layout (<= 64KB static) --------------------
struct Smem {
  float    sv[NE * NSV];                    // 16384 B : f32 master s_v / s_u,s_d
  alignas(16) _Float16 svh[NE * NSV];       //  8192 B : f16 snapshot of s_v (A frags)
  alignas(16) _Float16 pv[NE * NE * NPV];   // 16384 B : f16 master p_v
  alignas(16) _Float16 pblk[NE * NE * NPV]; // 16384 B : f16 A copy for p-GEMM / au|ad
  alignas(16) _Float16 msuh[NSV];           //   512 B : mean over up electrons
  alignas(16) _Float16 msdh[NSV];           //   512 B : mean over down electrons
  alignas(16) _Float16 mpuh[NE * NPV];      //  1024 B : mean_j<8  p_v  (per electron)
  alignas(16) _Float16 mpdh[NE * NPV];      //  1024 B : mean_j>=8 p_v
  float rs[NE * 3];                         //   192 B : r for this sample
  float as2[NA * 3];                        //    48 B : atom coords
  float env[NE];                            //    64 B : sum_a exp(-|r_i - a|)
  float rl[NE * NA];                        //   256 B : exp(-ra_len) scratch
  float orb[16 * 64];                       //  4096 B : 16 8x8 orbital matrices
  float sgn[16], ld[16];                    //   128 B : per-det sign / log|det|
};
// total ~65200 B

// ---------------- WMMA fragment helpers ------------------------------------
// A (16x32 f16): lane L holds row M=L%16.
//   lanes 0-15 : halves 0-7 -> K=k0..k0+7,   halves 8-15 -> K=k0+16..k0+23
//   lanes16-31 : halves 0-7 -> K=k0+8..k0+15,halves 8-15 -> K=k0+24..k0+31
DEVFN v16h load_a_frag(const _Float16* A, int lda, int kbase, int lane) {
  int m    = lane & 15;
  int koff = kbase + ((lane & 16) ? 8 : 0);
  const _Float16* p = A + m * lda + koff;
  v8h lo = *(const v8h*)(p);
  v8h hi = *(const v8h*)(p + 16);
  v16h r;
#pragma unroll
  for (int i = 0; i < 8; ++i) { r[i] = lo[i]; r[i + 8] = hi[i]; }
  return r;
}

// B (32x16 f16) from transposed weights Bt[N][Kpad]:
//   lane L -> column N = nbase + L%16; same K order as A:
//   lanes 0-15 : K=k0..k0+7, k0+16..k0+23; lanes 16-31 : K=k0+8..k0+15, k0+24..k0+31.
DEVFN v16h load_b_frag(const _Float16* Bt, int ldk, int nbase, int kbase, int lane) {
  int n    = nbase + (lane & 15);
  int koff = kbase + ((lane & 16) ? 8 : 0);
  const _Float16* p = Bt + (size_t)n * ldk + koff;
  v8h lo = *(const v8h*)(p);
  v8h hi = *(const v8h*)(p + 16);
  v16h r;
#pragma unroll
  for (int i = 0; i < 8; ++i) { r[i] = lo[i]; r[i + 8] = hi[i]; }
  return r;
}

DEVFN v8f wmma16x16x32(v16h a, v16h b, v8f c) {
  return __builtin_amdgcn_wmma_f32_16x16x32_f16(false, a, false, b, (short)0, c,
                                                false, false);
}

DEVFN v8f zero8f() {
  v8f z = {0.f, 0.f, 0.f, 0.f, 0.f, 0.f, 0.f, 0.f};
  return z;
}

// Composite A fragment for the 16x832 s_blk (never materialized):
//  kt 0-7  : s_v (f16 copy, 16x256)
//  kt 8-15 : m_su broadcast row (all M identical)
//  kt16-23 : m_sd broadcast row
//  kt 24   : m_pu per electron (16x32)
//  kt 25   : m_pd per electron
DEVFN v16h load_a_sblk(const Smem& s, int kt, int lane) {
  int m    = lane & 15;
  int koff = (lane & 16) ? 8 : 0;
  const _Float16* p;
  if (kt < 8)        p = s.svh  + m * NSV + kt * 32 + koff;
  else if (kt < 16)  p = s.msuh + (kt - 8) * 32 + koff;
  else if (kt < 24)  p = s.msdh + (kt - 16) * 32 + koff;
  else if (kt == 24) p = s.mpuh + m * 32 + koff;
  else               p = s.mpdh + m * 32 + koff;
  v8h lo = *(const v8h*)(p);
  v8h hi = *(const v8h*)(p + 16);
  v16h r;
#pragma unroll
  for (int i = 0; i < 8; ++i) { r[i] = lo[i]; r[i + 8] = hi[i]; }
  return r;
}

// C/D tile (16x16 f32): lane -> col N = lane%16, vgpr r -> row M = r + 8*(lane>=16)
// mode 0: sv = tanh(acc+b)        (layer 0, no residual)
// mode 1: sv += tanh(acc+b)       (residual layers)
// mode 2: rows <8 only  (s_u)     mode 3: rows >=8 only (s_d)
DEVFN void store_s(Smem& s, v8f acc, const float* bias, int nt, int lane, int mode) {
  int col   = nt * 16 + (lane & 15);
  float bv  = bias[col];
  int mbase = (lane & 16) ? 8 : 0;
#pragma unroll
  for (int rr = 0; rr < 8; ++rr) {
    int m   = rr + mbase;
    float v = fast_tanh(acc[rr] + bv);
    if (mode == 0)      s.sv[m * NSV + col] = v;
    else if (mode == 1) s.sv[m * NSV + col] += v;
    else if (mode == 2) { if (m < 8)  s.sv[m * NSV + col] = v; }
    else                { if (m >= 8) s.sv[m * NSV + col] = v; }
  }
}

// s GEMM over the composite 16x832 A, full N=256 (16 n-tiles over 8 waves)
DEVFN void smatmul_big(Smem& s, const _Float16* Bt, const float* bias, int mode, int tid) {
  int wave = tid >> 5, lane = tid & 31;
  int nt0 = wave * 2, nt1 = nt0 + 1;
  v8f acc0 = zero8f(), acc1 = zero8f();
  for (int kt = 0; kt < KT_BIG; ++kt) {
    v16h a  = load_a_sblk(s, kt, lane);
    v16h b0 = load_b_frag(Bt, KBIG, nt0 * 16, kt * 32, lane);
    v16h b1 = load_b_frag(Bt, KBIG, nt1 * 16, kt * 32, lane);
    acc0 = wmma16x16x32(a, b0, acc0);
    acc1 = wmma16x16x32(a, b1, acc1);
  }
  store_s(s, acc0, bias, nt0, lane, mode);
  store_s(s, acc1, bias, nt1, lane, mode);
}

// layer 0: A is 16x64 packed at the front of svh (lda=256), K padded 56->64
DEVFN void smatmul_l0(Smem& s, const _Float16* Bt, const float* bias, int tid) {
  int wave = tid >> 5, lane = tid & 31;
  int nt0 = wave * 2, nt1 = nt0 + 1;
  v8f acc0 = zero8f(), acc1 = zero8f();
#pragma unroll
  for (int kt = 0; kt < 2; ++kt) {
    v16h a  = load_a_frag(s.svh, NSV, kt * 32, lane);
    v16h b0 = load_b_frag(Bt, 64, nt0 * 16, kt * 32, lane);
    v16h b1 = load_b_frag(Bt, 64, nt1 * 16, kt * 32, lane);
    acc0 = wmma16x16x32(a, b0, acc0);
    acc1 = wmma16x16x32(a, b1, acc1);
  }
  store_s(s, acc0, bias, nt0, lane, 0);
  store_s(s, acc1, bias, nt1, lane, 0);
}

// p GEMM: A = pblk (256x32), B = Wt (32x32), 16 m-tiles x 2 n-tiles = 32 jobs
DEVFN void pmatmul(Smem& s, const _Float16* Wt, const float* bias, bool residual, int tid) {
  int wave = tid >> 5, lane = tid & 31;
  for (int job = wave; job < 32; job += 8) {
    int mt = job >> 1, nt = job & 1;
    v16h a = load_a_frag(s.pblk + mt * 16 * NPV, NPV, 0, lane);
    v16h b = load_b_frag(Wt, NPV, nt * 16, 0, lane);
    v8f acc = wmma16x16x32(a, b, zero8f());
    int col   = nt * 16 + (lane & 15);
    float bv  = bias[col];
    int mbase = (lane & 16) ? 8 : 0;
#pragma unroll
    for (int rr = 0; rr < 8; ++rr) {
      int row = mt * 16 + rr + mbase;
      float v = fast_tanh(acc[rr] + bv);
      if (residual) v += (float)s.pv[row * NPV + col];
      s.pv[row * NPV + col] = (_Float16)v;
    }
  }
}

// build f16 snapshot + means for the composite s_blk (layers >= 1 and final)
DEVFN void build_composite(Smem& s, int tid) {
  for (int idx = tid; idx < NE * NSV; idx += 256)
    s.svh[idx] = (_Float16)s.sv[idx];
  {
    int f = tid;  // exactly 256 threads
    float su = 0.f, sd = 0.f;
#pragma unroll
    for (int i = 0; i < 8; ++i) {
      su += s.sv[i * NSV + f];
      sd += s.sv[(i + 8) * NSV + f];
    }
    s.msuh[f] = (_Float16)(su * 0.125f);
    s.msdh[f] = (_Float16)(sd * 0.125f);
  }
  for (int idx = tid; idx < NE * NPV; idx += 256) {
    int i = idx >> 5, c = idx & 31;
    float pu = 0.f, pd = 0.f;
#pragma unroll
    for (int j = 0; j < 8; ++j) {
      pu += (float)s.pv[(i * 16 + j) * NPV + c];
      pd += (float)s.pv[(i * 16 + 8 + j) * NPV + c];
    }
    s.mpuh[idx] = (_Float16)(pu * 0.125f);
    s.mpdh[idx] = (_Float16)(pd * 0.125f);
  }
}

// ---------------------------------------------------------------------------
__global__ void __launch_bounds__(256)
wf_kernel(const float* __restrict__ r, const float* __restrict__ a,
          const _Float16* __restrict__ V0t, const _Float16* __restrict__ V1t,
          const _Float16* __restrict__ V2t, const _Float16* __restrict__ Vut,
          const _Float16* __restrict__ Vdt, const _Float16* __restrict__ W0t,
          const _Float16* __restrict__ W1t, const _Float16* __restrict__ W2t,
          const _Float16* __restrict__ wut, const _Float16* __restrict__ wdt,
          const float* __restrict__ V0b, const float* __restrict__ V1b,
          const float* __restrict__ V2b, const float* __restrict__ Vub,
          const float* __restrict__ Vdb, const float* __restrict__ W0b,
          const float* __restrict__ W1b, const float* __restrict__ W2b,
          const float* __restrict__ wub, const float* __restrict__ wdb,
          float* __restrict__ out) {
  __shared__ Smem s;
  const int tid = threadIdx.x;

  // ---------------- geometry ----------------
  if (tid < NE * 3) s.rs[tid]  = r[(size_t)blockIdx.x * NE * 3 + tid];
  if (tid < NA * 3) s.as2[tid] = a[tid];
  __syncthreads();

  if (tid < NE * NA) {   // electron-atom features -> sv cols 0..15
    int i = tid >> 2, at = tid & 3;
    float dx = s.rs[i * 3 + 0] - s.as2[at * 3 + 0];
    float dy = s.rs[i * 3 + 1] - s.as2[at * 3 + 1];
    float dz = s.rs[i * 3 + 2] - s.as2[at * 3 + 2];
    float len = sqrtf(dx * dx + dy * dy + dz * dz);
    s.sv[i * NSV + at * 4 + 0] = dx;
    s.sv[i * NSV + at * 4 + 1] = dy;
    s.sv[i * NSV + at * 4 + 2] = dz;
    s.sv[i * NSV + at * 4 + 3] = len;
    s.rl[tid] = fast_exp(-len);
  }
  {                       // electron-electron features -> pv cols 0..3
    int i = tid >> 4, j = tid & 15;
    float dx = s.rs[j * 3 + 0] - s.rs[i * 3 + 0];
    float dy = s.rs[j * 3 + 1] - s.rs[i * 3 + 1];
    float dz = s.rs[j * 3 + 2] - s.rs[i * 3 + 2];
    float ex = (i == j) ? 1.f : 0.f;   // eye adds 1 to all 3 comps on diagonal
    float lx = dx + ex, ly = dy + ex, lz = dz + ex;
    float len = sqrtf(lx * lx + ly * ly + lz * lz);
    int base = (i * 16 + j) * NPV;
    s.pv[base + 0] = (_Float16)dx;
    s.pv[base + 1] = (_Float16)dy;
    s.pv[base + 2] = (_Float16)dz;
    s.pv[base + 3] = (_Float16)len;
  }
  __syncthreads();
  if (tid < NE)
    s.env[tid] = s.rl[tid * 4] + s.rl[tid * 4 + 1] + s.rl[tid * 4 + 2] + s.rl[tid * 4 + 3];

  // ---------------- layer 0 (K = 56 padded to 64) ----------------
  if (tid < 16) {
    float su = 0.f, sd = 0.f;
#pragma unroll
    for (int i = 0; i < 8; ++i) {
      su += s.sv[i * NSV + tid];
      sd += s.sv[(i + 8) * NSV + tid];
    }
    s.msuh[tid] = (_Float16)(su * 0.125f);
    s.msdh[tid] = (_Float16)(sd * 0.125f);
  }
  if (tid < 64) {
    int i = tid >> 2, c = tid & 3;
    float pu = 0.f, pd = 0.f;
#pragma unroll
    for (int j = 0; j < 8; ++j) {
      pu += (float)s.pv[(i * 16 + j) * NPV + c];
      pd += (float)s.pv[(i * 16 + 8 + j) * NPV + c];
    }
    s.mpuh[tid] = (_Float16)(pu * 0.125f);
    s.mpdh[tid] = (_Float16)(pd * 0.125f);
  }
  __syncthreads();
  for (int idx = tid; idx < NE * 64; idx += 256) {  // pack 16x64 A into svh
    int i = idx >> 6, f = idx & 63;
    float v;
    if (f < 16)      v = s.sv[i * NSV + f];
    else if (f < 32) v = (float)s.msuh[f - 16];
    else if (f < 48) v = (float)s.msdh[f - 32];
    else if (f < 52) v = (float)s.mpuh[i * 4 + (f - 48)];
    else if (f < 56) v = (float)s.mpdh[i * 4 + (f - 52)];
    else             v = 0.f;
    s.svh[i * NSV + f] = (_Float16)v;
  }
  __syncthreads();
  smatmul_l0(s, V0t, V0b, tid);           // s_v = tanh(s_blk @ V0 + b)
  __syncthreads();
  for (int idx = tid; idx < NE * NE * NPV; idx += 256) {  // pad K 4 -> 32
    int c = idx & 31;
    s.pblk[idx] = (c < 4) ? s.pv[idx] : (_Float16)0.f;
  }
  __syncthreads();
  pmatmul(s, W0t, W0b, false, tid);       // p_v = tanh(p_v @ W0 + b)
  __syncthreads();

  // ---------------- layers 1..2 (residual) ----------------
  const _Float16* Vtl[2] = {V1t, V2t};
  const float*    Vbl[2] = {V1b, V2b};
  const _Float16* Wtl[2] = {W1t, W2t};
  const float*    Wbl[2] = {W1b, W2b};
  for (int l = 0; l < 2; ++l) {
    build_composite(s, tid);
    __syncthreads();
    smatmul_big(s, Vtl[l], Vbl[l], 1, tid);   // s_v += tanh(...)
    __syncthreads();
    for (int idx = tid; idx < NE * NE * NPV; idx += 256) s.pblk[idx] = s.pv[idx];
    __syncthreads();
    pmatmul(s, Wtl[l], Wbl[l], true, tid);    // p_v += tanh(...)
    __syncthreads();
  }

  // ---------------- final s_blk, Vu / Vd heads ----------------
  build_composite(s, tid);
  __syncthreads();
  smatmul_big(s, Vut, Vub, 2, tid);  // s_u -> sv rows 0..7
  smatmul_big(s, Vdt, Vdb, 3, tid);  // s_d -> sv rows 8..15
  __syncthreads();

  // au (rows 0..7 = s_u) | ad (rows 0..7 = s_d), zero padded rows 8..15
  for (int idx = tid; idx < 2 * NE * NSV; idx += 256) {
    int buf = idx >> 12;            // 0 = au, 1 = ad
    int rem = idx & 4095;
    int m = rem >> 8, k = rem & 255;
    float v = (m < 8) ? s.sv[(buf ? (8 + m) : m) * NSV + k] : 0.f;
    s.pblk[idx] = (_Float16)v;
  }
  __syncthreads();

  // orbital heads: waves 0-3 -> wu (dets 0..7), waves 4-7 -> wd (dets 8..15)
  {
    int wave = tid >> 5, lane = tid & 31;
    const _Float16* Abuf = (wave < 4) ? s.pblk : (s.pblk + NE * NSV);
    const _Float16* Wt   = (wave < 4) ? wut : wdt;
    const float*    bb   = (wave < 4) ? wub : wdb;
    int nt      = wave & 3;
    int envoff  = (wave < 4) ? 0 : 8;
    int orbbase = (wave < 4) ? 0 : 8;
    v8f acc = zero8f();
#pragma unroll
    for (int kt = 0; kt < 8; ++kt) {
      v16h a = load_a_frag(Abuf, NSV, kt * 32, lane);
      v16h b = load_b_frag(Wt, NSV, nt * 16, kt * 32, lane);
      acc = wmma16x16x32(a, b, acc);
    }
    int col = nt * 16 + (lane & 15);         // 0..63 -> (det, orbital)
    int d = col >> 3, o = col & 7;
    float bv = bb[col];
    int mbase = (lane & 16) ? 8 : 0;
#pragma unroll
    for (int rr = 0; rr < 8; ++rr) {
      int m = rr + mbase;
      if (m < 8) {
        float val = (acc[rr] + bv) * s.env[envoff + m];
        s.orb[(orbbase + d) * 64 + m * 8 + o] = val;
      }
    }
  }
  __syncthreads();

  // ---------------- slogdet of 16 8x8 matrices (pivoted LU) ----------------
  if (tid < 16) {
    float* M = s.orb + tid * 64;
    float sign = 1.f, logd = 0.f;
    for (int k = 0; k < 8; ++k) {
      int p = k;
      float mx = fabsf(M[k * 8 + k]);
      for (int rr = k + 1; rr < 8; ++rr) {
        float av = fabsf(M[rr * 8 + k]);
        if (av > mx) { mx = av; p = rr; }
      }
      if (p != k) {
        for (int c = 0; c < 8; ++c) {
          float t = M[k * 8 + c]; M[k * 8 + c] = M[p * 8 + c]; M[p * 8 + c] = t;
        }
        sign = -sign;
      }
      float piv = M[k * 8 + k];
      if (piv < 0.f) sign = -sign;
      logd += fast_log(fabsf(piv));
      float inv = __builtin_amdgcn_rcpf(piv);
      for (int rr = k + 1; rr < 8; ++rr) {
        float fct = M[rr * 8 + k] * inv;
        for (int c = k + 1; c < 8; ++c) M[rr * 8 + c] -= fct * M[k * 8 + c];
      }
    }
    s.sgn[tid] = sign;
    s.ld[tid]  = logd;
  }
  __syncthreads();

  if (tid == 0) {  // signed logsumexp over 8 determinant products
    float lg[8], sg[8], mmax = -1e30f;
#pragma unroll
    for (int d = 0; d < 8; ++d) {
      lg[d] = s.ld[d] + s.ld[8 + d];
      sg[d] = s.sgn[d] * s.sgn[8 + d];
      mmax = fmaxf(mmax, lg[d]);
    }
    float ssum = 0.f;
#pragma unroll
    for (int d = 0; d < 8; ++d) ssum += sg[d] * fast_exp(lg[d] - mmax);
    out[blockIdx.x] = fast_log(fabsf(ssum)) + mmax;
  }
}

// ---------------------------------------------------------------------------
// Weight prep: fp32 (K x N) -> f16 transposed (N x Kpad), zero pad K -> Kpad.
__global__ void conv_tr_kernel(const float* __restrict__ W, _Float16* __restrict__ Wt,
                               int K, int N, int Kpad) {
  int idx = blockIdx.x * blockDim.x + threadIdx.x;
  int total = N * Kpad;
  if (idx >= total) return;
  int n = idx / Kpad, k = idx - n * Kpad;
  Wt[idx] = (k < K) ? (_Float16)W[k * N + n] : (_Float16)0.f;
}

// ---------------------------------------------------------------------------
extern "C" void kernel_launch(void* const* d_in, const int* in_sizes, int n_in,
                              void* d_out, int out_size, void* d_ws, size_t ws_size,
                              hipStream_t stream) {
  const float* r   = (const float*)d_in[0];
  const float* a   = (const float*)d_in[1];
  const float* V0w = (const float*)d_in[2];  const float* V0b = (const float*)d_in[3];
  const float* W0w = (const float*)d_in[4];  const float* W0b = (const float*)d_in[5];
  const float* V1w = (const float*)d_in[6];  const float* V1b = (const float*)d_in[7];
  const float* W1w = (const float*)d_in[8];  const float* W1b = (const float*)d_in[9];
  const float* V2w = (const float*)d_in[10]; const float* V2b = (const float*)d_in[11];
  const float* W2w = (const float*)d_in[12]; const float* W2b = (const float*)d_in[13];
  const float* Vuw = (const float*)d_in[14]; const float* Vub = (const float*)d_in[15];
  const float* Vdw = (const float*)d_in[16]; const float* Vdb = (const float*)d_in[17];
  const float* wuw = (const float*)d_in[18]; const float* wub = (const float*)d_in[19];
  const float* wdw = (const float*)d_in[20]; const float* wdb = (const float*)d_in[21];

  _Float16* ws  = (_Float16*)d_ws;
  _Float16* V0t = ws;
  _Float16* V1t = V0t + 256 * 64;
  _Float16* V2t = V1t + 256 * 832;
  _Float16* Vut = V2t + 256 * 832;
  _Float16* Vdt = Vut + 256 * 832;
  _Float16* W0t = Vdt + 256 * 832;
  _Float16* W1t = W0t + 32 * 32;
  _Float16* W2t = W1t + 32 * 32;
  _Float16* wut = W2t + 32 * 32;
  _Float16* wdt = wut + 64 * 256;

  auto tr = [&](const float* W, _Float16* Wt, int K, int N, int Kpad) {
    int total = N * Kpad;
    conv_tr_kernel<<<(total + 255) / 256, 256, 0, stream>>>(W, Wt, K, N, Kpad);
  };
  tr(V0w, V0t, 56, 256, 64);
  tr(V1w, V1t, 832, 256, 832);
  tr(V2w, V2t, 832, 256, 832);
  tr(Vuw, Vut, 832, 256, 832);
  tr(Vdw, Vdt, 832, 256, 832);
  tr(W0w, W0t, 4, 32, 32);
  tr(W1w, W1t, 32, 32, 32);
  tr(W2w, W2t, 32, 32, 32);
  tr(wuw, wut, 256, 64, 256);
  tr(wdw, wdt, 256, 64, 256);

  int Bn = in_sizes[0] / (NE * 3);
  wf_kernel<<<Bn, 256, 0, stream>>>(r, a, V0t, V1t, V2t, Vut, Vdt,
                                    W0t, W1t, W2t, wut, wdt,
                                    V0b, V1b, V2b, Vub, Vdb,
                                    W0b, W1b, W2b, wub, wdb,
                                    (float*)d_out);
}
